// Net_40372692582720
// MI455X (gfx1250) — compile-verified, matrix operand order checked by reading, not run on hardware
//
#include <hip/hip_runtime.h>
#include <hip/hip_bf16.h>
#include <math.h>

typedef __bf16 bf16;
typedef __attribute__((ext_vector_type(16))) __bf16 v16bf;
typedef __attribute__((ext_vector_type(8)))  __bf16 v8bf;
typedef __attribute__((ext_vector_type(8)))  float  v8f;

__device__ __forceinline__ float sigmoidf_(float x){ return 1.0f/(1.0f+__expf(-x)); }

// monotonic float<->uint encoding for atomicMax-based segment max
__device__ __forceinline__ unsigned fenc(float f){
  int i = __float_as_int(f);
  return (unsigned)(i ^ ((i>>31) | (int)0x80000000));
}
__device__ __forceinline__ float fdec(unsigned k){
  int i = (k & 0x80000000u) ? (int)(k & 0x7fffffffu) : ~(int)k;
  return __int_as_float(i);
}

// ---------------- small utility kernels ----------------

__global__ void k_f32_to_bf16(const float* __restrict__ in, bf16* __restrict__ out, int n){
  int i = blockIdx.x*blockDim.x + threadIdx.x;
  if (i < n) out[i] = (bf16)in[i];
}

__global__ void k_fill_u32(unsigned* __restrict__ p, unsigned v, int n){
  int i = blockIdx.x*blockDim.x + threadIdx.x;
  if (i < n) p[i] = v;
}

// H1 = relu(edge_attr @ nn1_w^T + nn1_b) -> bf16, shape (E,128)
__global__ __launch_bounds__(128) void k_edge_mlp1(
    const float* __restrict__ ea, const float* __restrict__ w1,
    const float* __restrict__ b1, bf16* __restrict__ h1, int E){
  int e = blockIdx.x; int j = threadIdx.x;
  if (e >= E) return;
  const float* a = ea + (size_t)e*5;
  float acc = b1[j];
#pragma unroll
  for (int t=0;t<5;++t) acc += a[t]*w1[j*5+t];
  h1[(size_t)e*128 + j] = (bf16)fmaxf(acc, 0.0f);
}

__global__ void k_deg(const int* __restrict__ dst, float* __restrict__ deg, int E){
  int e = blockIdx.x*blockDim.x + threadIdx.x;
  if (e < E) atomicAdd(&deg[dst[e]], 1.0f);
}
__global__ void k_invdeg(float* __restrict__ deg, int N){
  int n = blockIdx.x*blockDim.x + threadIdx.x;
  if (n < N) deg[n] = 1.0f / fmaxf(deg[n], 1.0f);
}

// out = h = relu(x @ lin0_w^T + lin0_b)
__global__ __launch_bounds__(64) void k_lin0(
    const float* __restrict__ x, const float* __restrict__ w,
    const float* __restrict__ b, float* __restrict__ outv, float* __restrict__ h,
    int N, int F){
  int n = blockIdx.x; int o = threadIdx.x;
  if (n >= N) return;
  float acc = b[o];
  for (int t=0;t<F;++t) acc += x[(size_t)n*F+t]*w[o*F+t];
  acc = fmaxf(acc, 0.0f);
  outv[(size_t)n*64+o] = acc;
  h[(size_t)n*64+o] = acc;
}

// ---------------- fused NNConv message kernel (WMMA) ----------------
// Per ISA 7.12.2 (16-bit A 16x32): lane<16 -> row=lane&15, K halves {0..7,16..23};
// lane>=16 -> K halves {8..15,24..31}. B assumed symmetric over N. Row stride 128 bf16.
__device__ __forceinline__ v16bf load_frag(const bf16* __restrict__ base, int row, int k0, int lane){
  const bf16* p = base + (size_t)row*128 + k0 + ((lane & 16) ? 8 : 0);
  v8bf lo = *(const v8bf*)(p);
  v8bf hi = *(const v8bf*)(p + 16);
  v16bf f;
#pragma unroll
  for (int t=0;t<8;++t){ f[t]=lo[t]; f[t+8]=hi[t]; }
  return f;
}

__global__ __launch_bounds__(256) void k_nnconv_msg(
    const bf16* __restrict__ h1, const bf16* __restrict__ w2,
    const float* __restrict__ b2, const float* __restrict__ outv,
    const int* __restrict__ src, const int* __restrict__ dst,
    float* __restrict__ agg, int E)
{
  __shared__ float s_out[32][64];   // out[src] for 32 edges
  __shared__ float s_msg[32][64];   // accumulated messages
  __shared__ float s_b2[4096];      // nn2_b
  __shared__ int   s_src[32], s_dst[32];
  int tid = threadIdx.x;
  int e0  = blockIdx.x * 32;
  if (tid < 32){
    int e = e0 + tid;
    s_src[tid] = (e < E) ? src[e] : -1;
    s_dst[tid] = (e < E) ? dst[e] : -1;
  }
  for (int idx=tid; idx<2048; idx+=256) (&s_msg[0][0])[idx] = 0.0f;
  for (int idx=tid; idx<4096; idx+=256) s_b2[idx] = b2[idx];
  __syncthreads();
  for (int idx=tid; idx<2048; idx+=256){
    int el = idx>>6, k = idx&63;
    int s = s_src[el];
    (&s_out[0][0])[idx] = (s >= 0) ? outv[(size_t)s*64+k] : 0.0f;
  }
  __syncthreads();

  int wave = tid>>5, lane = tid&31;
  int o0    = (wave & 3) * 16;     // output-column range of this wave
  int ibase = (wave >> 2) * 32;    // i-half of this wave
  int lrow  = lane & 15;
  int lhi   = (lane & 16) ? 8 : 0;

  // A fragments: H1 rows for 32 edges (2 M-tiles), all 4 K-steps, held in VGPRs
  v16bf afrag[2][4];
#pragma unroll
  for (int m=0;m<2;++m){
    int r = e0 + m*16 + lrow;
    if (r >= E) r = E-1;
#pragma unroll
    for (int ks=0;ks<4;++ks) afrag[m][ks] = load_frag(h1, r, ks*32, lane);
  }
  float msgacc[2][8];
#pragma unroll
  for (int m=0;m<2;++m)
#pragma unroll
    for (int v=0;v<8;++v) msgacc[m][v]=0.0f;

  for (int ii=0; ii<32; ++ii){
    int i  = ibase + ii;
    int c0 = i*64 + o0;            // flat We column base (i*DIM + o)
    v16bf bfrag[4];
#pragma unroll
    for (int ks=0;ks<4;++ks) bfrag[ks] = load_frag(w2, c0 + lrow, ks*32, lane);
    float bias = s_b2[c0 + lrow];
#pragma unroll
    for (int m=0;m<2;++m){
      v8f acc;
#pragma unroll
      for (int v=0;v<8;++v) acc[v]=bias;
#pragma unroll
      for (int ks=0;ks<4;++ks)
        acc = __builtin_amdgcn_wmma_f32_16x16x32_bf16(
                false, afrag[m][ks], false, bfrag[ks], (short)0, acc, false, false);
      // D element (M=v+lhi, N=lrow) == We[edge=m*16+v+lhi][i][o0+lrow]
      int er = m*16 + lhi;
#pragma unroll
      for (int v=0;v<8;++v)
        msgacc[m][v] += s_out[er+v][i] * acc[v];
    }
  }
  int ocol = o0 + lrow;
#pragma unroll
  for (int m=0;m<2;++m){
    int er = m*16 + lhi;
#pragma unroll
    for (int v=0;v<8;++v)
      atomicAdd(&s_msg[er+v][ocol], msgacc[m][v]);   // combine the two i-half waves
  }
  __syncthreads();
  for (int idx=tid; idx<2048; idx+=256){
    int el = idx>>6, o = idx&63;
    int d = s_dst[el];
    if (d >= 0) atomicAdd(&agg[(size_t)d*64+o], s_msg[el][o]);
  }
}

// ---------------- fused root GEMV + GRU step ----------------
__global__ __launch_bounds__(256) void k_node_update(
    const float* __restrict__ agg, const float* __restrict__ invdeg,
    float* __restrict__ outv, float* __restrict__ h,
    const float* __restrict__ root_w, const float* __restrict__ conv_b,
    const float* __restrict__ gw_ih, const float* __restrict__ gw_hh,
    const float* __restrict__ gb_ih, const float* __restrict__ gb_hh, int N)
{
  __shared__ float s_rootT[64*64];    // [k][o]
  __shared__ float s_wihT[64*192];    // [k][j]
  __shared__ float s_whhT[64*192];    // [k][j]
  __shared__ float s_cb[64];
  __shared__ float s_bih[192];
  __shared__ float s_bhh[192];
  __shared__ float s_vec[8][3][64];   // [wave][0=out,1=h,2=m][k]
  int tid = threadIdx.x;
  for (int idx=tid; idx<4096; idx+=256){ int o=idx&63, k=idx>>6; s_rootT[k*64+o] = root_w[o*64+k]; }
  for (int idx=tid; idx<12288; idx+=256){
    int k = idx/192, j = idx%192;
    s_wihT[k*192+j] = gw_ih[j*64+k];
    s_whhT[k*192+j] = gw_hh[j*64+k];
  }
  if (tid < 64)  s_cb[tid]  = conv_b[tid];
  if (tid < 192){ s_bih[tid] = gb_ih[tid]; s_bhh[tid] = gb_hh[tid]; }
  __syncthreads();

  int wave = tid>>5, lane = tid&31;
  for (int t=0;t<8;++t){
    int node = blockIdx.x*64 + t*8 + wave;
    if (node >= N) continue;
    float inv = invdeg[node];
    const float* ag = agg + (size_t)node*64;
    float* op = outv + (size_t)node*64;
    float* hp = h    + (size_t)node*64;
    float o0v = op[lane], o1v = op[lane+32];
    float h0  = hp[lane], h1v = hp[lane+32];
    s_vec[wave][0][lane]=o0v; s_vec[wave][0][lane+32]=o1v;
    s_vec[wave][1][lane]=h0;  s_vec[wave][1][lane+32]=h1v;

    float a0 = s_cb[lane]    + ag[lane]*inv;
    float a1 = s_cb[lane+32] + ag[lane+32]*inv;
#pragma unroll 4
    for (int k=0;k<64;++k){
      float ok = s_vec[wave][0][k];
      a0 += ok * s_rootT[k*64+lane];
      a1 += ok * s_rootT[k*64+lane+32];
    }
    float m0 = fmaxf(a0,0.0f), m1 = fmaxf(a1,0.0f);
    s_vec[wave][2][lane]=m0; s_vec[wave][2][lane+32]=m1;

    float ai[6], ah[6];
#pragma unroll
    for (int u=0;u<6;++u){ ai[u]=s_bih[lane+32*u]; ah[u]=s_bhh[lane+32*u]; }
#pragma unroll 2
    for (int k=0;k<64;++k){
      float mk = s_vec[wave][2][k];
      float hk = s_vec[wave][1][k];
      const float* wi = s_wihT + k*192 + lane;
      const float* wh = s_whhT + k*192 + lane;
#pragma unroll
      for (int u=0;u<6;++u){ ai[u] += mk*wi[32*u]; ah[u] += hk*wh[32*u]; }
    }
    { // o = lane   (r:t0, z:t2, n:t4)
      float r = sigmoidf_(ai[0]+ah[0]);
      float z = sigmoidf_(ai[2]+ah[2]);
      float nn = tanhf(ai[4] + r*ah[4]);
      float hn = (1.0f-z)*nn + z*h0;
      hp[lane]=hn; op[lane]=hn;
    }
    { // o = lane+32 (r:t1, z:t3, n:t5)
      float r = sigmoidf_(ai[1]+ah[1]);
      float z = sigmoidf_(ai[3]+ah[3]);
      float nn = tanhf(ai[5] + r*ah[5]);
      float hn = (1.0f-z)*nn + z*h1v;
      hp[lane+32]=hn; op[lane+32]=hn;
    }
  }
}

// ---------------- Set2Set ----------------
__global__ __launch_bounds__(256) void k_lstm(
    const float* __restrict__ qstar, float* __restrict__ hs, float* __restrict__ cs,
    const float* __restrict__ w_ih, const float* __restrict__ w_hh,
    const float* __restrict__ b_ih, const float* __restrict__ b_hh)
{
  int g = blockIdx.x; int tid = threadIdx.x;
  __shared__ float s_q[128]; __shared__ float s_h[64]; __shared__ float s_g[256];
  if (tid < 128) s_q[tid] = qstar[(size_t)g*128+tid];
  if (tid >= 128 && tid < 192) s_h[tid-128] = hs[(size_t)g*64 + tid-128];
  __syncthreads();
  float acc = b_ih[tid] + b_hh[tid];
  for (int k=0;k<128;++k) acc += s_q[k]*w_ih[tid*128+k];
  for (int k=0;k<64;++k)  acc += s_h[k]*w_hh[tid*64+k];
  s_g[tid] = acc;
  __syncthreads();
  if (tid < 64){
    float ig = sigmoidf_(s_g[tid]);
    float fg = sigmoidf_(s_g[64+tid]);
    float gg = tanhf(s_g[128+tid]);
    float og = sigmoidf_(s_g[192+tid]);
    float c = fg*cs[(size_t)g*64+tid] + ig*gg;
    cs[(size_t)g*64+tid] = c;
    hs[(size_t)g*64+tid] = og*tanhf(c);
  }
}

__global__ __launch_bounds__(256) void k_attn_e(
    const float* __restrict__ outv, const float* __restrict__ hs,
    const int* __restrict__ batch, float* __restrict__ ebuf,
    unsigned* __restrict__ emax, int N){
  int wave = threadIdx.x>>5, lane = threadIdx.x&31;
  int n = blockIdx.x*8 + wave;
  if (n >= N) return;
  int g = batch[n];
  const float* q  = hs + (size_t)g*64;
  const float* ov = outv + (size_t)n*64;
  float p = ov[lane]*q[lane] + ov[lane+32]*q[lane+32];
#pragma unroll
  for (int off=16; off; off>>=1) p += __shfl_xor(p, off, 32);
  if (lane == 0){ ebuf[n] = p; atomicMax(&emax[g], fenc(p)); }
}

__global__ void k_attn_a(const float* __restrict__ ebuf, const int* __restrict__ batch,
    const unsigned* __restrict__ emax, float* __restrict__ abuf, float* __restrict__ asum, int N){
  int n = blockIdx.x*blockDim.x + threadIdx.x;
  if (n >= N) return;
  int g = batch[n];
  float a = __expf(ebuf[n] - fdec(emax[g]));
  abuf[n] = a;
  atomicAdd(&asum[g], a);
}

__global__ __launch_bounds__(256) void k_attn_r(
    const float* __restrict__ outv, const float* __restrict__ abuf,
    const float* __restrict__ asum, const int* __restrict__ batch,
    float* __restrict__ rbuf, int N){
  int wave = threadIdx.x>>5, lane = threadIdx.x&31;
  int n = blockIdx.x*8 + wave;
  if (n >= N) return;
  int g = batch[n];
  float c = abuf[n] / asum[g];
  atomicAdd(&rbuf[(size_t)g*64+lane],    c*outv[(size_t)n*64+lane]);
  atomicAdd(&rbuf[(size_t)g*64+lane+32], c*outv[(size_t)n*64+lane+32]);
}

__global__ void k_qstar(const float* __restrict__ hs, const float* __restrict__ rbuf,
                        float* __restrict__ qstar, int B){
  int i = blockIdx.x*blockDim.x + threadIdx.x;
  if (i >= B*128) return;
  int g = i>>7, k = i&127;
  qstar[i] = (k < 64) ? hs[(size_t)g*64+k] : rbuf[(size_t)g*64 + (k-64)];
}

__global__ __launch_bounds__(64) void k_readout(
    const float* __restrict__ qs, const float* __restrict__ w1, const float* __restrict__ b1,
    const float* __restrict__ w2, const float* __restrict__ b2f, float* __restrict__ y){
  int g = blockIdx.x; int o = threadIdx.x;
  __shared__ float s[64];
  float acc = b1[o];
  const float* q = qs + (size_t)g*128;
  for (int k=0;k<128;++k) acc += q[k]*w1[o*128+k];
  s[o] = fmaxf(acc, 0.0f)*w2[o];
  __syncthreads();
  if (o < 32){
    float v = s[o] + s[o+32];
#pragma unroll
    for (int off=16; off; off>>=1) v += __shfl_xor(v, off, 32);
    if (o == 0) y[g] = v + b2f[0];
  }
}

// ---------------- launch ----------------
extern "C" void kernel_launch(void* const* d_in, const int* in_sizes, int n_in,
                              void* d_out, int out_size, void* d_ws, size_t ws_size,
                              hipStream_t stream){
  const float* x      = (const float*)d_in[0];
  const int*   eidx   = (const int*)  d_in[1];
  const float* eattr  = (const float*)d_in[2];
  const int*   batch  = (const int*)  d_in[3];
  const float* lin0_w = (const float*)d_in[4];  const float* lin0_b = (const float*)d_in[5];
  const float* nn1_w  = (const float*)d_in[6];  const float* nn1_b  = (const float*)d_in[7];
  const float* nn2_w  = (const float*)d_in[8];  const float* nn2_b  = (const float*)d_in[9];
  const float* root_w = (const float*)d_in[10]; const float* conv_b = (const float*)d_in[11];
  const float* gw_ih  = (const float*)d_in[12]; const float* gw_hh  = (const float*)d_in[13];
  const float* gb_ih  = (const float*)d_in[14]; const float* gb_hh  = (const float*)d_in[15];
  const float* lw_ih  = (const float*)d_in[16]; const float* lw_hh  = (const float*)d_in[17];
  const float* lb_ih  = (const float*)d_in[18]; const float* lb_hh  = (const float*)d_in[19];
  const float* lin1_w = (const float*)d_in[20]; const float* lin1_b = (const float*)d_in[21];
  const float* lin2_w = (const float*)d_in[22]; const float* lin2_b = (const float*)d_in[23];

  const int N = in_sizes[3];
  const int E = in_sizes[2] / 5;
  const int F = in_sizes[0] / N;
  const int B = out_size;
  const int* src = eidx;
  const int* dst = eidx + E;

  char* w = (char*)d_ws;
  auto alloc = [&](size_t bytes)->char*{
    char* p = w; w += (bytes + 255) & ~(size_t)255; return p;
  };
  bf16*     h1     = (bf16*)    alloc((size_t)E*128*2);
  bf16*     w2b    = (bf16*)    alloc((size_t)4096*128*2);
  float*    outv   = (float*)   alloc((size_t)N*64*4);
  float*    h      = (float*)   alloc((size_t)N*64*4);
  float*    agg    = (float*)   alloc((size_t)N*64*4);
  float*    invdeg = (float*)   alloc((size_t)N*4);
  float*    ebuf   = (float*)   alloc((size_t)N*4);
  float*    abuf   = (float*)   alloc((size_t)N*4);
  unsigned* emax   = (unsigned*)alloc((size_t)B*4);
  float*    asum   = (float*)   alloc((size_t)B*4);
  float*    rbuf   = (float*)   alloc((size_t)B*64*4);
  float*    qstar  = (float*)   alloc((size_t)B*128*4);
  float*    hs     = (float*)   alloc((size_t)B*64*4);
  float*    cs     = (float*)   alloc((size_t)B*64*4);
  (void)ws_size; (void)n_in;

  // init
  hipMemsetAsync(invdeg, 0, (size_t)N*4, stream);
  hipMemsetAsync(qstar,  0, (size_t)B*128*4, stream);
  hipMemsetAsync(hs,     0, (size_t)B*64*4, stream);
  hipMemsetAsync(cs,     0, (size_t)B*64*4, stream);

  k_f32_to_bf16<<<(4096*128+255)/256, 256, 0, stream>>>(nn2_w, w2b, 4096*128);
  k_edge_mlp1  <<<E, 128, 0, stream>>>(eattr, nn1_w, nn1_b, h1, E);
  k_deg        <<<(E+255)/256, 256, 0, stream>>>(dst, invdeg, E);
  k_invdeg     <<<(N+255)/256, 256, 0, stream>>>(invdeg, N);
  k_lin0       <<<N, 64, 0, stream>>>(x, lin0_w, lin0_b, outv, h, N, F);

  for (int it=0; it<3; ++it){
    hipMemsetAsync(agg, 0, (size_t)N*64*4, stream);
    k_nnconv_msg  <<<(E+31)/32, 256, 0, stream>>>(h1, w2b, nn2_b, outv, src, dst, agg, E);
    k_node_update <<<(N+63)/64, 256, 0, stream>>>(agg, invdeg, outv, h, root_w, conv_b,
                                                  gw_ih, gw_hh, gb_ih, gb_hh, N);
  }

  for (int s2=0; s2<3; ++s2){
    k_lstm     <<<B, 256, 0, stream>>>(qstar, hs, cs, lw_ih, lw_hh, lb_ih, lb_hh);
    k_fill_u32 <<<(B+255)/256, 256, 0, stream>>>(emax, 0x00800000u /* fenc(-FLT_MAX) */, B);
    hipMemsetAsync(asum, 0, (size_t)B*4, stream);
    hipMemsetAsync(rbuf, 0, (size_t)B*64*4, stream);
    k_attn_e   <<<(N+7)/8,   256, 0, stream>>>(outv, hs, batch, ebuf, emax, N);
    k_attn_a   <<<(N+255)/256,256,0, stream>>>(ebuf, batch, emax, abuf, asum, N);
    k_attn_r   <<<(N+7)/8,   256, 0, stream>>>(outv, abuf, asum, batch, rbuf, N);
    k_qstar    <<<(B*128+255)/256, 256, 0, stream>>>(hs, rbuf, qstar, B);
  }
  k_readout<<<B, 64, 0, stream>>>(qstar, lin1_w, lin1_b, lin2_w, lin2_b, (float*)d_out);
}
